// VectorizedMoE_54193897341571
// MI455X (gfx1250) — matrix operand order repaired and checked by reading, not compiled, not run on hardware
//
#include <hip/hip_runtime.h>
#include <hip/hip_bf16.h>

// ---------------------------------------------------------------------------
// MoE forward for MI455X (gfx1250): bf16 WMMA FFN, async LDS staging,
// deterministic routing. B=4,S=2048,H=1024 ; E=8,K=2,I=4096 ; capacity=1280
// ---------------------------------------------------------------------------

typedef __attribute__((ext_vector_type(16))) __bf16 v16bf;
typedef __attribute__((ext_vector_type(8)))  __bf16 v8bf;
typedef __attribute__((ext_vector_type(8)))  float  v8f;

#define T_TOK 8192
#define HDIM  1024
#define EXP   8
#define IDIM  4096
#define CAP   1280

#if defined(__gfx1250__) && __has_builtin(__builtin_amdgcn_global_load_async_to_lds_b128)
#define HAVE_ASYNC_LDS 1
#else
#define HAVE_ASYNC_LDS 0
#endif

// Exact pointee type from the builtin's signature: GCC-vector int4.
typedef int v4i_vs __attribute__((vector_size(16)));
typedef __attribute__((address_space(1))) v4i_vs as1_v4i;
typedef __attribute__((address_space(3))) v4i_vs as3_v4i;

static __device__ __forceinline__ void async_cp16(const void* g, void* l) {
#if HAVE_ASYNC_LDS
  // generic->AS1: flat == global numerically; generic->AS3: low 32 bits of a
  // flat LDS address are the LDS byte offset (ISA aperture model).
  __builtin_amdgcn_global_load_async_to_lds_b128(
      reinterpret_cast<as1_v4i*>((size_t)g),
      reinterpret_cast<as3_v4i*>((size_t)l), 0, 0);
#else
  (void)g; (void)l;
#endif
}

static __device__ __forceinline__ void wait_async0() {
#if defined(__gfx1250__)
#if __has_builtin(__builtin_amdgcn_s_wait_asynccnt)
  __builtin_amdgcn_s_wait_asynccnt(0);
#else
  asm volatile("s_wait_asynccnt 0" ::: "memory");
#endif
#endif
}

static __device__ __forceinline__ __bf16 f2bf(float f) {
  unsigned u = __builtin_bit_cast(unsigned, f);
  unsigned r = u + 0x7FFFu + ((u >> 16) & 1u);      // round-to-nearest-even
  unsigned short h = (unsigned short)(r >> 16);
  return __builtin_bit_cast(__bf16, h);
}

static __device__ __forceinline__ unsigned pack2(float lo, float hi) {
  unsigned l  = (unsigned)__builtin_bit_cast(unsigned short, f2bf(lo));
  unsigned hh = (unsigned)__builtin_bit_cast(unsigned short, f2bf(hi));
  return l | (hh << 16);
}

// ---------------------------------------------------------------------------
// 0) Streaming fp32 -> bf16 weight conversion (8 elems / thread)
// ---------------------------------------------------------------------------
__global__ __launch_bounds__(256) void moe_cvt_kernel(
    const float* __restrict__ src, __bf16* __restrict__ dst)
{
  int i = blockIdx.x * 256 + threadIdx.x;
  const float4* s = (const float4*)src + (size_t)i * 2;
  float4 a = s[0], b = s[1];
  uint4 w = make_uint4(pack2(a.x, a.y), pack2(a.z, a.w),
                       pack2(b.x, b.y), pack2(b.z, b.w));
  *((uint4*)dst + (size_t)i) = w;
}

// ---------------------------------------------------------------------------
// 1) Router: one wave32 per token. dot[e] = <x_t, emb_e>, softmax, top-2.
// ---------------------------------------------------------------------------
__global__ __launch_bounds__(256) void moe_router_kernel(
    const float* __restrict__ x, const float* __restrict__ emb,
    int2* __restrict__ top2_idx, float2* __restrict__ top2_val,
    float* __restrict__ partial_soft, float* __restrict__ partial_cnt)
{
  const int lane = threadIdx.x & 31;
  const int wave = threadIdx.x >> 5;
  const int tok  = blockIdx.x * 8 + wave;

  __shared__ float sprob[8][EXP];
  __shared__ float scnt[8][EXP];

  float part[EXP];
#pragma unroll
  for (int e = 0; e < EXP; ++e) part[e] = 0.f;

  const float* xr = x + (size_t)tok * HDIM;
  for (int i = 0; i < HDIM / 32; ++i) {
    float xv = xr[lane + 32 * i];
#pragma unroll
    for (int e = 0; e < EXP; ++e)
      part[e] += xv * emb[e * HDIM + lane + 32 * i];
  }
#pragma unroll
  for (int e = 0; e < EXP; ++e) {
    float v = part[e];
#pragma unroll
    for (int off = 16; off; off >>= 1) v += __shfl_xor(v, off, 32);
    part[e] = v;
  }

  if (lane == 0) {
    float m = part[0];
#pragma unroll
    for (int e = 1; e < EXP; ++e) m = fmaxf(m, part[e]);
    float p[EXP]; float s = 0.f;
#pragma unroll
    for (int e = 0; e < EXP; ++e) { p[e] = __expf(part[e] - m); s += p[e]; }
    float inv = 1.f / s;
#pragma unroll
    for (int e = 0; e < EXP; ++e) p[e] *= inv;
    int i0 = 0;
    for (int e = 1; e < EXP; ++e) if (p[e] > p[i0]) i0 = e;   // ties: lowest idx
    int i1 = -1;
    for (int e = 0; e < EXP; ++e) {
      if (e == i0) continue;
      if (i1 < 0 || p[e] > p[i1]) i1 = e;
    }
    top2_idx[tok] = make_int2(i0, i1);
    top2_val[tok] = make_float2(p[i0], p[i1]);
    for (int e = 0; e < EXP; ++e) {
      sprob[wave][e] = p[e];
      scnt[wave][e]  = (e == i0 || e == i1) ? 1.f : 0.f;
    }
  }
  __syncthreads();
  if (threadIdx.x < EXP) {
    int e = threadIdx.x;
    float ss = 0.f, cc = 0.f;
    for (int w = 0; w < 8; ++w) { ss += sprob[w][e]; cc += scnt[w][e]; }
    partial_soft[blockIdx.x * EXP + e] = ss;
    partial_cnt[blockIdx.x * EXP + e]  = cc;
  }
}

// ---------------------------------------------------------------------------
// 2) Loss reduction (single block, fixed order -> deterministic)
// ---------------------------------------------------------------------------
__global__ void moe_loss_kernel(const float* __restrict__ partial_soft,
                                const float* __restrict__ partial_cnt,
                                float* __restrict__ out_loss)
{
  __shared__ float contrib[EXP];
  if (threadIdx.x < EXP) {
    int e = threadIdx.x;
    float ss = 0.f, cc = 0.f;
    for (int b = 0; b < T_TOK / 8; ++b) {
      ss += partial_soft[b * EXP + e];
      cc += partial_cnt[b * EXP + e];
    }
    contrib[e] = (cc / (float)T_TOK) * ss;
  }
  __syncthreads();
  if (threadIdx.x == 0) {
    float l = 0.f;
    for (int e = 0; e < EXP; ++e) l += contrib[e];
    out_loss[0] = l * ((float)EXP / (float)T_TOK);
  }
}

// ---------------------------------------------------------------------------
// 3) Capacity: wave e scans tokens in order with ballot prefix-sum.
// ---------------------------------------------------------------------------
__global__ __launch_bounds__(256) void moe_capacity_kernel(
    const int2* __restrict__ top2_idx,
    int* __restrict__ slot_token, int* __restrict__ token_slot)
{
  const int lane = threadIdx.x & 31;
  const int e    = threadIdx.x >> 5;

  for (int i = lane; i < CAP; i += 32) slot_token[e * CAP + i] = -1;

  int base = 0;
  for (int c = 0; c < T_TOK / 32; ++c) {
    int t = c * 32 + lane;
    int2 ti = top2_idx[t];
    bool m = (ti.x == e) || (ti.y == e);
    unsigned mask = (unsigned)__ballot(m);
    int prefix = __popc(mask & ((1u << lane) - 1u));
    if (m) {
      int slot = base + prefix;
      int j = (ti.x == e) ? 0 : 1;
      if (slot < CAP) {
        slot_token[e * CAP + slot] = t;
        token_slot[t * 2 + j] = slot;
      } else {
        token_slot[t * 2 + j] = -1;
      }
    }
    base += __popc(mask);
  }
}

// ---------------------------------------------------------------------------
// 4) Gather tokens into [E*CAP, H] bf16 (zero-fill unused slots), 128b ops
// ---------------------------------------------------------------------------
__global__ __launch_bounds__(256) void moe_gather_kernel(
    const float* __restrict__ x, const int* __restrict__ slot_token,
    __bf16* __restrict__ tokens)
{
  int row = blockIdx.x;
  int tok = slot_token[row];
  __bf16* dst = tokens + (size_t)row * HDIM;
  int h = threadIdx.x * 4;                 // 256 threads x 4 elems = 1024
  if (tok < 0) {
    *(uint2*)(dst + h) = make_uint2(0u, 0u);
  } else {
    float4 v = *(const float4*)(x + (size_t)tok * HDIM + h);
    *(uint2*)(dst + h) = make_uint2(pack2(v.x, v.y), pack2(v.z, v.w));
  }
}

// ---------------------------------------------------------------------------
// 5/6) Tiled bf16 WMMA GEMM:  C[m,n] = sum_k A[m,k] * Bw[n,k]
//   A: [E][CAP][KD] bf16.  Bw: [E][ND][KD] bf16 (row n contiguous in k).
//   Block tile 128x128, 8 waves (4x2), wave tile 32x64 = 2x4 WMMA accums.
//   Double-buffered LDS stage (K=32), async global->LDS when available.
// ---------------------------------------------------------------------------
template <int KD, int ND, bool RELU_BF16>
__global__ __launch_bounds__(256) void moe_gemm_kernel(
    const __bf16* __restrict__ Aall, const __bf16* __restrict__ Ball,
    void* __restrict__ OutAll)
{
  constexpr int LDK = 40;                       // 32 k-halfs + pad (80 B rows)
  __shared__ __bf16 As[2][128][LDK];
  __shared__ __bf16 Bs[2][128][LDK];

  const int e  = blockIdx.z;
  const int m0 = blockIdx.y * 128;
  const int n0 = blockIdx.x * 128;

  const __bf16* A = Aall + (size_t)e * CAP * KD + (size_t)m0 * KD;
  const __bf16* B = Ball + (size_t)e * ND  * KD + (size_t)n0 * KD;

  const int tid  = threadIdx.x;
  const int lane = tid & 31;
  const int wave = tid >> 5;
  const int wm = wave & 3, wn = wave >> 2;
  const int lrow = lane & 15;
  const int akb = (lane < 16) ? 0 : 8;    // A frag K base (ISA 16-bit A layout)
  const int bkb = (lane < 16) ? 0 : 8;    // B frag K base (ISA 16-bit B layout)

  const int srow = tid >> 1;              // staging row/col 0..127
  const int scol = (tid & 1) * 16;        // staging k-chunk (halfs)

  v8f acc[2][4];
#pragma unroll
  for (int mi = 0; mi < 2; ++mi)
#pragma unroll
    for (int ni = 0; ni < 4; ++ni) {
      v8f z = {};
      acc[mi][ni] = z;
    }

  const size_t srow_off = (size_t)srow * KD + scol;

  auto compute_stage = [&](int buf) {
    v16bf af[2], bf_[4];
#pragma unroll
    for (int mi = 0; mi < 2; ++mi) {
      v8bf lo = *(const v8bf*)&As[buf][wm * 32 + mi * 16 + lrow][akb];
      v8bf hi = *(const v8bf*)&As[buf][wm * 32 + mi * 16 + lrow][akb + 16];
      af[mi] = __builtin_shufflevector(lo, hi, 0, 1, 2, 3, 4, 5, 6, 7,
                                       8, 9, 10, 11, 12, 13, 14, 15);
    }
#pragma unroll
    for (int ni = 0; ni < 4; ++ni) {
      v8bf lo = *(const v8bf*)&Bs[buf][wn * 64 + ni * 16 + lrow][bkb];
      v8bf hi = *(const v8bf*)&Bs[buf][wn * 64 + ni * 16 + lrow][bkb + 16];
      bf_[ni] = __builtin_shufflevector(lo, hi, 0, 1, 2, 3, 4, 5, 6, 7,
                                        8, 9, 10, 11, 12, 13, 14, 15);
    }
#pragma unroll
    for (int mi = 0; mi < 2; ++mi)
#pragma unroll
      for (int ni = 0; ni < 4; ++ni) {
#if defined(__gfx1250__)
        acc[mi][ni] = __builtin_amdgcn_wmma_f32_16x16x32_bf16(
            false, af[mi], false, bf_[ni], (short)0, acc[mi][ni],
            false, false);
#else
        acc[mi][ni] = acc[mi][ni];  // host-pass placeholder
#endif
      }
  };

  constexpr int NST = KD / 32;

#if HAVE_ASYNC_LDS
  auto issue_stage = [&](int s, int buf) {
    const __bf16* ag = A + srow_off + s * 32;
    const __bf16* bg = B + srow_off + s * 32;
#pragma unroll
    for (int j = 0; j < 2; ++j) {
      async_cp16(ag + j * 8, &As[buf][srow][scol + j * 8]);
      async_cp16(bg + j * 8, &Bs[buf][srow][scol + j * 8]);
    }
  };

  issue_stage(0, 0);
  wait_async0();
  __syncthreads();
  for (int s = 0; s < NST; ++s) {
    int cur = s & 1;
    if (s + 1 < NST) issue_stage(s + 1, 1 - cur);   // overlaps with WMMA below
    compute_stage(cur);
    wait_async0();
    __syncthreads();
  }
#else
  uint4 pa[2], pb[2];
  auto load_stage = [&](int s) {
    const uint4* ag = (const uint4*)(A + srow_off + s * 32);
    const uint4* bg = (const uint4*)(B + srow_off + s * 32);
#pragma unroll
    for (int j = 0; j < 2; ++j) { pa[j] = ag[j]; pb[j] = bg[j]; }
  };
  auto store_stage = [&](int buf) {
#pragma unroll
    for (int j = 0; j < 2; ++j) {
      *(uint4*)&As[buf][srow][scol + j * 8] = pa[j];
      *(uint4*)&Bs[buf][srow][scol + j * 8] = pb[j];
    }
  };

  load_stage(0);
  store_stage(0);
  __syncthreads();
  for (int s = 0; s < NST; ++s) {
    int cur = s & 1;
    if (s + 1 < NST) load_stage(s + 1);             // overlaps with WMMA below
    compute_stage(cur);
    if (s + 1 < NST) store_stage(1 - cur);
    __syncthreads();
  }
#endif

  // ---- epilogue: C layout = lane (0-15:n, +16:+8 rows), vgpr r = row ----
#pragma unroll
  for (int mi = 0; mi < 2; ++mi) {
    int rb = m0 + wm * 32 + mi * 16 + (lane >> 4) * 8;
#pragma unroll
    for (int ni = 0; ni < 4; ++ni) {
      int col = n0 + wn * 64 + ni * 16 + lrow;
      if (RELU_BF16) {
        __bf16* Out = (__bf16*)OutAll + (size_t)e * CAP * ND;
#pragma unroll
        for (int r = 0; r < 8; ++r) {
          float v = acc[mi][ni][r];
          v = v > 0.f ? v : 0.f;
          Out[(size_t)(rb + r) * ND + col] = f2bf(v);
        }
      } else {
        float* Out = (float*)OutAll + (size_t)e * CAP * ND;
#pragma unroll
        for (int r = 0; r < 8; ++r)
          Out[(size_t)(rb + r) * ND + col] = acc[mi][ni][r];
      }
    }
  }
}

// ---------------------------------------------------------------------------
// 7) Final gather: out[t] = sum_j w_j * result[e_j, slot_j, :]  (128b ops)
// ---------------------------------------------------------------------------
__global__ __launch_bounds__(256) void moe_final_kernel(
    const float* __restrict__ result, const int2* __restrict__ top2_idx,
    const float2* __restrict__ top2_val, const int* __restrict__ token_slot,
    float* __restrict__ out)
{
  int t = blockIdx.x;
  int2 ti = top2_idx[t];
  float2 tv = top2_val[t];
  int s0 = token_slot[t * 2 + 0];
  int s1 = token_slot[t * 2 + 1];
  float4 a = make_float4(0.f, 0.f, 0.f, 0.f);
  if (s0 >= 0) {
    float4 v = ((const float4*)(result + (size_t)(ti.x * CAP + s0) * HDIM))[threadIdx.x];
    a.x += tv.x * v.x; a.y += tv.x * v.y; a.z += tv.x * v.z; a.w += tv.x * v.w;
  }
  if (s1 >= 0) {
    float4 v = ((const float4*)(result + (size_t)(ti.y * CAP + s1) * HDIM))[threadIdx.x];
    a.x += tv.y * v.x; a.y += tv.y * v.y; a.z += tv.y * v.z; a.w += tv.y * v.w;
  }
  ((float4*)(out + (size_t)t * HDIM))[threadIdx.x] = a;
}

// ---------------------------------------------------------------------------
extern "C" void kernel_launch(void* const* d_in, const int* in_sizes, int n_in,
                              void* d_out, int out_size, void* d_ws, size_t ws_size,
                              hipStream_t stream) {
  const float* x   = (const float*)d_in[0];  // [T, H]
  const float* emb = (const float*)d_in[1];  // [E, H]
  const float* w1  = (const float*)d_in[2];  // [E, I, H]
  const float* w2  = (const float*)d_in[3];  // [E, H, I]
  float* out = (float*)d_out;                // [T*H] final + [1] loss

  char* ws = (char*)d_ws;
  size_t off = 0;
  auto take = [&](size_t bytes) {
    size_t o = off;
    off = (off + bytes + 255) & ~(size_t)255;
    return o;
  };
  int2*   top2_idx     = (int2*)(ws + take((size_t)T_TOK * sizeof(int2)));
  float2* top2_val     = (float2*)(ws + take((size_t)T_TOK * sizeof(float2)));
  float*  partial_soft = (float*)(ws + take((size_t)(T_TOK / 8) * EXP * 4));
  float*  partial_cnt  = (float*)(ws + take((size_t)(T_TOK / 8) * EXP * 4));
  int*    slot_token   = (int*)(ws + take((size_t)EXP * CAP * 4));
  int*    token_slot   = (int*)(ws + take((size_t)T_TOK * 2 * 4));
  __bf16* tokens_bf    = (__bf16*)(ws + take((size_t)EXP * CAP * HDIM * 2));
  __bf16* inter_bf     = (__bf16*)(ws + take((size_t)EXP * CAP * IDIM * 2));
  float*  result_f     = (float*)(ws + take((size_t)EXP * CAP * HDIM * 4));
  __bf16* w1_bf        = (__bf16*)(ws + take((size_t)EXP * IDIM * HDIM * 2));
  __bf16* w2_bf        = (__bf16*)(ws + take((size_t)EXP * HDIM * IDIM * 2));
  (void)ws_size; (void)in_sizes; (void)n_in; (void)out_size;

  const int welems = EXP * IDIM * HDIM;      // 33,554,432
  moe_cvt_kernel<<<welems / (256 * 8), 256, 0, stream>>>(w1, w1_bf);
  moe_cvt_kernel<<<welems / (256 * 8), 256, 0, stream>>>(w2, w2_bf);

  moe_router_kernel<<<T_TOK / 8, 256, 0, stream>>>(
      x, emb, top2_idx, top2_val, partial_soft, partial_cnt);

  moe_loss_kernel<<<1, 64, 0, stream>>>(
      partial_soft, partial_cnt, out + (size_t)T_TOK * HDIM);

  moe_capacity_kernel<<<1, 256, 0, stream>>>(top2_idx, slot_token, token_slot);

  moe_gather_kernel<<<EXP * CAP, 256, 0, stream>>>(x, slot_token, tokens_bf);

  // GEMM1: [CAP x IDIM] = tokens[CAP x H] * W1^T, relu, bf16 out
  moe_gemm_kernel<HDIM, IDIM, true>
      <<<dim3(IDIM / 128, CAP / 128, EXP), 256, 0, stream>>>(
          tokens_bf, w1_bf, (void*)inter_bf);

  // GEMM2: [CAP x H] = inter[CAP x IDIM] * W2^T, fp32 out
  moe_gemm_kernel<IDIM, HDIM, false>
      <<<dim3(HDIM / 128, CAP / 128, EXP), 256, 0, stream>>>(
          inter_bf, w2_bf, (void*)result_f);

  moe_final_kernel<<<T_TOK, 256, 0, stream>>>(
      result_f, top2_idx, top2_val, token_slot, out);
}
